// Network_2499670966368
// MI455X (gfx1250) — matrix operand order repaired and checked by reading, not compiled, not run on hardware
//
#include <hip/hip_runtime.h>

// ---------------------------------------------------------------------------
// Types for CDNA5 WMMA (wave32) and TDM descriptors
// ---------------------------------------------------------------------------
typedef __attribute__((ext_vector_type(16))) __bf16 v16bf;
typedef __attribute__((ext_vector_type(8)))  __bf16 v8bf;
typedef __attribute__((ext_vector_type(8)))  float  v8f;
typedef __attribute__((ext_vector_type(4)))  unsigned int u32x4;
typedef __attribute__((ext_vector_type(8)))  int i32x8;
typedef __attribute__((ext_vector_type(4)))  int i32x4;

#define D128 128
#define NATOMS 4096
#define NRES 512
#define KSLICES 4

#if __has_builtin(__builtin_amdgcn_tensor_load_to_lds) && \
    __has_builtin(__builtin_amdgcn_s_wait_tensorcnt)
#define HAVE_TDM 1
#endif

// float -> bf16 round-to-nearest-even via bit ops (no reliance on cvt support)
__device__ __forceinline__ __bf16 f2b(float x) {
  unsigned u = __builtin_bit_cast(unsigned, x);
  unsigned r = (u + 0x7FFFu + ((u >> 16) & 1u)) >> 16;
  unsigned short s = (unsigned short)r;
  return __builtin_bit_cast(__bf16, s);
}

__device__ __forceinline__ float sigmf(float x) { return 1.0f / (1.0f + expf(-x)); }

// ---------------------------------------------------------------------------
// Elementwise helpers
// ---------------------------------------------------------------------------
__global__ __launch_bounds__(256) void f2bf_kernel(const float* __restrict__ src,
                                                   __bf16* __restrict__ dst, int n) {
  int i = blockIdx.x * 256 + threadIdx.x;
  if (i < n) dst[i] = f2b(src[i]);
}

// dst(C x R) bf16 = transpose of src(R x C) f32
__global__ __launch_bounds__(256) void tconv_kernel(const float* __restrict__ src,
                                                    __bf16* __restrict__ dst, int R, int C) {
  int i = blockIdx.x * 256 + threadIdx.x;
  if (i < R * C) {
    int r = i / C, c = i % C;
    dst[c * R + r] = f2b(src[i]);
  }
}

__global__ __launch_bounds__(256) void embed_kernel(const float* __restrict__ emb,
                                                    const int* __restrict__ atoms,
                                                    float* __restrict__ H,
                                                    __bf16* __restrict__ Hbf) {
  int i = blockIdx.x * 256 + threadIdx.x;   // NATOMS*128 threads exactly
  int r = i >> 7, d = i & 127;
  float v = emb[atoms[r] * D128 + d];
  H[i] = v;
  Hbf[i] = f2b(v);
}

// Fixed-order reduction of KSLICES split-K partials -> bf16 (deterministic)
__global__ __launch_bounds__(256) void reduce4_bf_kernel(const float* __restrict__ P,
                                                         __bf16* __restrict__ dst, int n) {
  int i = blockIdx.x * 256 + threadIdx.x;
  if (i < n) {
    float s = ((P[i] + P[(size_t)n + i]) + P[2 * (size_t)n + i]) + P[3 * (size_t)n + i];
    dst[i] = f2b(s);
  }
}

// GRU cell update (in-place on H), also refresh bf16 copy for next GEMM step.
__global__ __launch_bounds__(256) void gru_kernel(const float* __restrict__ GI,
                                                  const float* __restrict__ GH,
                                                  float* __restrict__ H,
                                                  __bf16* __restrict__ Hbf) {
  int i = blockIdx.x * 256 + threadIdx.x;   // NATOMS*128 threads
  int r = i >> 7, d = i & 127;
  const float* gi = GI + (size_t)r * 384;
  const float* gh = GH + (size_t)r * 384;
  float rg = sigmf(gi[d] + gh[d]);
  float z  = sigmf(gi[128 + d] + gh[128 + d]);
  float n  = tanhf(gi[256 + d] + rg * gh[256 + d]);
  float h  = (1.0f - z) * n + z * H[i];
  H[i] = h;
  Hbf[i] = f2b(h);
}

// ---------------------------------------------------------------------------
// TDM: stage a (rows x 128) bf16 panel from global into LDS (D# per ISA ch.8).
// 2D tensor, data_size code 1 (2 bytes), type=2 ("image"), groups 2/3 zero.
// ---------------------------------------------------------------------------
#ifdef HAVE_TDM
__device__ __forceinline__ void tdm_load_panel(const __bf16* gsrc, unsigned ldsOff,
                                               int rows, int cols) {
  unsigned long long ga = (unsigned long long)(size_t)gsrc;
  u32x4 g0;
  g0[0] = 1u;                                            // count=1, user descriptor
  g0[1] = ldsOff;                                        // lds_addr (bytes)
  g0[2] = (unsigned)(ga & 0xFFFFFFFFu);                  // global_addr[31:0]
  g0[3] = (unsigned)((ga >> 32) & 0x01FFFFFFu) | (2u << 30);  // addr[56:32] | type=2
  i32x8 g1;
  g1[0] = (1 << 16);                                     // workgroup_mask=0, data_size=1 (2B)
  g1[1] = (cols & 0xFFFF) << 16;                         // tensor_dim0 lo16 @ bit48
  g1[2] = ((cols >> 16) & 0xFFFF) | ((rows & 0xFFFF) << 16);  // dim0 hi | tensor_dim1 lo
  g1[3] = ((rows >> 16) & 0xFFFF) | ((cols & 0xFFFF) << 16);  // dim1 hi | tile_dim0
  g1[4] = (rows & 0xFFFF);                               // tile_dim1 ; tile_dim2=0
  g1[5] = cols;                                          // tensor_dim0_stride[31:0]
  g1[6] = 0;                                             // stride hi | dim1_stride lo
  g1[7] = 0;
  i32x4 z4 = {0, 0, 0, 0};
#if defined(__clang_major__) && (__clang_major__ >= 23)
  i32x8 z8 = {0, 0, 0, 0, 0, 0, 0, 0};
  __builtin_amdgcn_tensor_load_to_lds(g0, g1, z4, z4, z8, 0);
#else
  __builtin_amdgcn_tensor_load_to_lds(g0, g1, z4, z4, 0);
#endif
}
#endif

// ---------------------------------------------------------------------------
// Split-K strip GEMM for the dominant product: Cpart[s] = A[:,Ks] * B[Ks,:]
// A: M x K bf16 (adjacency), B: K x 128 bf16 (h), Cpart: KSLICES x M x 128 f32.
// One wave owns a 16x128 strip: one A fragment feeds 8 WMMAs (8x A reuse).
// B k-panels (32x128 bf16 = 8KB) are DOUBLE-BUFFERED in LDS via TDM: the DMA
// of panel k+1 is issued before computing panel k, and TENSORcnt is drained
// only at the iteration-end barrier, hiding the transfer behind 8 WMMAs.
// blockIdx.x = strip group, blockIdx.y = K slice.
// ---------------------------------------------------------------------------
__global__ __launch_bounds__(256) void wmma_strip_gemm_kernel(
    const __bf16* __restrict__ A, const __bf16* __restrict__ B,
    float* __restrict__ Cpart, int M, int K) {
  __shared__ __align__(16) __bf16 Bpanel[2][32 * 128];
  const int lane = threadIdx.x & 31;
  const int wave = threadIdx.x >> 5;
  const int slice = blockIdx.y;
  const int strip = blockIdx.x * 8 + wave;
  const int m0 = strip << 4;
  const int hi = lane >> 4, lm = lane & 15;
  const int kLen = K / KSLICES;
  const int kBeg = slice * kLen;
  const int kEnd = kBeg + kLen;

  v8f acc[8];
#pragma unroll
  for (int nt = 0; nt < 8; ++nt)
#pragma unroll
    for (int e = 0; e < 8; ++e) acc[nt][e] = 0.0f;

  const __bf16* Arow = A + (size_t)(m0 + lm) * K + hi * 8;

  union V16 { v16bf v; v8bf h[2]; };

  // ---- prologue: stage the first panel ----
#ifdef HAVE_TDM
  if (wave == 0) {
    tdm_load_panel(B + (size_t)kBeg * 128, (unsigned)(size_t)(void*)&Bpanel[0][0], 32, 128);
    __builtin_amdgcn_s_wait_tensorcnt(0);
  }
#else
  {
    const __bf16* src = B + (size_t)kBeg * 128;
    int t = threadIdx.x;
    *(v8bf*)(&Bpanel[0][0] + t * 16)     = *(const v8bf*)(src + t * 16);
    *(v8bf*)(&Bpanel[0][0] + t * 16 + 8) = *(const v8bf*)(src + t * 16 + 8);
  }
#endif
  __syncthreads();

  int buf = 0;
  for (int k0 = kBeg; k0 < kEnd; k0 += 32) {
    const bool hasNext = (k0 + 32) < kEnd;
    // issue DMA for next panel into the other buffer (consumed next iteration)
#ifdef HAVE_TDM
    if (hasNext && wave == 0)
      tdm_load_panel(B + (size_t)(k0 + 32) * 128,
                     (unsigned)(size_t)(void*)&Bpanel[buf ^ 1][0], 32, 128);
#else
    if (hasNext) {
      const __bf16* src = B + (size_t)(k0 + 32) * 128;
      int t = threadIdx.x;
      *(v8bf*)(&Bpanel[buf ^ 1][0] + t * 16)     = *(const v8bf*)(src + t * 16);
      *(v8bf*)(&Bpanel[buf ^ 1][0] + t * 16 + 8) = *(const v8bf*)(src + t * 16 + 8);
    }
#endif
    // compute on current panel
    const __bf16* Brow = &Bpanel[buf][(hi * 8) * 128 + lm];
    V16 a;
    a.h[0] = *(const v8bf*)(Arow + k0);        // K = k0 + hi*8 + 0..7
    a.h[1] = *(const v8bf*)(Arow + k0 + 16);   // K = k0 + 16 + hi*8 + 0..7
#pragma unroll
    for (int nt = 0; nt < 8; ++nt) {
      V16 b;
#pragma unroll
      for (int s = 0; s < 8; ++s) {
        b.v[s]     = Brow[s * 128 + nt * 16];
        b.v[s + 8] = Brow[(s + 16) * 128 + nt * 16];
      }
      acc[nt] = __builtin_amdgcn_wmma_f32_16x16x32_bf16(
          false, a.v, false, b.v, (short)0, acc[nt], false, false);
    }
#ifdef HAVE_TDM
    if (hasNext && wave == 0) __builtin_amdgcn_s_wait_tensorcnt(0);
#endif
    __syncthreads();           // next panel landed & all consumers done
    buf ^= 1;
  }

  float* Cp = Cpart + ((size_t)slice * M + m0 + hi * 8) * 128 + lm;
#pragma unroll
  for (int nt = 0; nt < 8; ++nt)
#pragma unroll
    for (int r = 0; r < 8; ++r) Cp[(size_t)r * 128 + nt * 16] = acc[nt][r];
}

// ---------------------------------------------------------------------------
// Generic bf16 WMMA GEMM for the smaller products (one wave per 16x16 tile).
// Fragment layouts per CDNA5 ISA 7.12.2 (wave32).
// ---------------------------------------------------------------------------
__global__ __launch_bounds__(256) void wmma_gemm_kernel(const __bf16* __restrict__ A,
                                                        const __bf16* __restrict__ B,
                                                        const float* __restrict__ bias,
                                                        float* __restrict__ C,
                                                        int M, int N, int K) {
  const int lane = threadIdx.x & 31;
  const int wave = threadIdx.x >> 5;
  const int tile = blockIdx.x * 8 + wave;
  const int ntN  = N >> 4;
  const int mt   = tile / ntN;
  const int nt   = tile % ntN;
  if (mt >= (M >> 4)) return;           // wave-uniform: EXEC stays all-ones
  const int m0 = mt << 4, n0 = nt << 4;
  const int hi = lane >> 4, lm = lane & 15;

  v8f acc;
#pragma unroll
  for (int e = 0; e < 8; ++e) acc[e] = 0.0f;

  const __bf16* Arow  = A + (size_t)(m0 + lm) * K + hi * 8;
  const __bf16* Bbase = B + (size_t)(hi * 8) * N + n0 + lm;

  union V16 { v16bf v; v8bf h[2]; };

  for (int k0 = 0; k0 < K; k0 += 32) {
    V16 a, b;
    a.h[0] = *(const v8bf*)(Arow + k0);
    a.h[1] = *(const v8bf*)(Arow + k0 + 16);
    const __bf16* bp = Bbase + (size_t)k0 * N;
#pragma unroll
    for (int s = 0; s < 8; ++s) {
      b.v[s]     = bp[(size_t)s * N];
      b.v[s + 8] = bp[(size_t)(s + 16) * N];
    }
    acc = __builtin_amdgcn_wmma_f32_16x16x32_bf16(
        false, a.v, false, b.v, (short)0, acc, false, false);
  }

  const int n = n0 + lm;
  const float bv = bias ? bias[n] : 0.0f;
  float* Crow = C + (size_t)(m0 + hi * 8) * N + n;
#pragma unroll
  for (int r = 0; r < 8; ++r) Crow[(size_t)r * N] = acc[r] + bv;
}

// ---------------------------------------------------------------------------
// conv1 via exact factorization.  With f = (c<<18)|(y<<9)|x the reference's
// reshape gives X(c,y,x) = 0.5*HF[4c + (y>>7)][x&127]
//                        + 0.5*HF[(4y + (x>>7)) & 511][x&127].
// The channel sum of the first term depends only on (tap t, q=y>>7, m=x&127):
//   P[t][q][m][oc] = 0.5 * sum_c w[oc][c][t] * HF[4c+q][m]
//   Q[t][oc]       = 0.5 * sum_c w[oc][c][t]
// => out(oc,y,x) = b[oc] + sum_t [in-bounds] (P[t][q][m][oc] + Q[t][oc]*hr)
// with hr = HF[(4yy+(xx>>7))&511][xx&127].  270 FLOP/pixel vs 17280.
// ---------------------------------------------------------------------------
__global__ __launch_bounds__(256) void conv1_pq_kernel(const float* __restrict__ HF,
                                                       const float* __restrict__ W,
                                                       float* __restrict__ P,
                                                       float* __restrict__ Q) {
  int i = blockIdx.x * 256 + threadIdx.x;           // 9*4*128*16 = 73728 threads
  if (i >= 9 * 4 * 128 * 16) return;
  int oc = i & 15;
  int m  = (i >> 4) & 127;
  int q  = (i >> 11) & 3;
  int t  = i >> 13;
  float s = 0.0f;
  if (oc < 15) {
    for (int c = 0; c < 128; ++c)
      s = fmaf(W[((size_t)oc * 128 + c) * 9 + t], HF[(4 * c + q) * 128 + m], s);
    s *= 0.5f;
  }
  P[i] = s;
  if (q == 0 && m == 0) {
    float qs = 0.0f;
    if (oc < 15) {
      for (int c = 0; c < 128; ++c) qs += W[((size_t)oc * 128 + c) * 9 + t];
      qs *= 0.5f;
    }
    Q[t * 16 + oc] = qs;
  }
}

__global__ __launch_bounds__(256) void conv1_fast_kernel(const float* __restrict__ HF,
                                                         const float* __restrict__ P,
                                                         const float* __restrict__ Q,
                                                         const float* __restrict__ Bc,
                                                         float* __restrict__ Out) {
  const int x = (blockIdx.x << 4) + (threadIdx.x & 15);
  const int y = (blockIdx.y << 4) + (threadIdx.x >> 4);
  float acc[15];
#pragma unroll
  for (int oc = 0; oc < 15; ++oc) acc[oc] = Bc[oc];
#pragma unroll
  for (int t = 0; t < 9; ++t) {
    int yy = y + (t / 3) - 1, xx = x + (t % 3) - 1;
    if (yy < 0 || yy >= NRES || xx < 0 || xx >= NRES) continue;
    int q = yy >> 7, m = xx & 127;
    float hr = HF[(((yy << 2) + (xx >> 7)) & 511) * 128 + m];
    const float* p  = P + ((((t * 4 + q) * 128) + m) << 4);
    const float* qq = Q + t * 16;
#pragma unroll
    for (int oc = 0; oc < 15; ++oc) acc[oc] += p[oc] + qq[oc] * hr;
  }
#pragma unroll
  for (int oc = 0; oc < 15; ++oc)
    Out[((size_t)oc << 18) + ((size_t)y << 9) + x] = fmaxf(acc[oc], 0.0f);
}

// ---------------------------------------------------------------------------
// conv2/conv3: direct 3x3 conv, weights staged in LDS
// ---------------------------------------------------------------------------
template <int IC, int OC, bool RELU>
__global__ __launch_bounds__(256) void convK_kernel(const float* __restrict__ In,
                                                    const float* __restrict__ W,
                                                    const float* __restrict__ Bc,
                                                    float* __restrict__ Out) {
  __shared__ float wl[OC * IC * 9];
  for (int i = threadIdx.x; i < OC * IC * 9; i += 256) wl[i] = W[i];
  __syncthreads();
  const int x = (blockIdx.x << 4) + (threadIdx.x & 15);
  const int y = (blockIdx.y << 4) + (threadIdx.x >> 4);
  float acc[OC];
#pragma unroll
  for (int oc = 0; oc < OC; ++oc) acc[oc] = Bc[oc];
  for (int ic = 0; ic < IC; ++ic) {
    float xv[9];
#pragma unroll
    for (int t = 0; t < 9; ++t) {
      int yy = y + (t / 3) - 1, xx = x + (t % 3) - 1;
      xv[t] = (yy >= 0 && yy < NRES && xx >= 0 && xx < NRES)
                  ? In[((size_t)ic << 18) + ((size_t)yy << 9) + xx] : 0.0f;
    }
#pragma unroll
    for (int oc = 0; oc < OC; ++oc) {
      const float* wp = &wl[(oc * IC + ic) * 9];
#pragma unroll
      for (int t = 0; t < 9; ++t) acc[oc] = fmaf(xv[t], wp[t], acc[oc]);
    }
  }
#pragma unroll
  for (int oc = 0; oc < OC; ++oc) {
    float v = acc[oc];
    if (RELU) v = fmaxf(v, 0.0f);
    Out[((size_t)oc << 18) + ((size_t)y << 9) + x] = v;
  }
}

__global__ __launch_bounds__(256) void sym_sigmoid_kernel(const float* __restrict__ C3,
                                                          float* __restrict__ out, int n) {
  int i = blockIdx.x * 256 + threadIdx.x;
  if (i >= n) return;
  int o = i >> 18, y = (i >> 9) & 511, x = i & 511;
  float a = C3[i];
  float b = C3[((size_t)o << 18) + ((size_t)x << 9) + y];
  out[i] = sigmf(0.5f * (a + b));
}

// ---------------------------------------------------------------------------
// Orchestration
// ---------------------------------------------------------------------------
extern "C" void kernel_launch(void* const* d_in, const int* in_sizes, int n_in,
                              void* d_out, int out_size, void* d_ws, size_t ws_size,
                              hipStream_t stream) {
  const float* adjacency = (const float*)d_in[0];
  const float* membership= (const float*)d_in[1];
  const float* emb       = (const float*)d_in[2];
  const float* lin_w     = (const float*)d_in[3];
  const float* lin_b     = (const float*)d_in[4];
  const float* w_ih      = (const float*)d_in[5];
  const float* w_hh      = (const float*)d_in[6];
  const float* b_ih      = (const float*)d_in[7];
  const float* b_hh      = (const float*)d_in[8];
  const float* c1w       = (const float*)d_in[9];
  const float* c1b       = (const float*)d_in[10];
  const float* c2w       = (const float*)d_in[11];
  const float* c2b       = (const float*)d_in[12];
  const float* c3w       = (const float*)d_in[13];
  const float* c3b       = (const float*)d_in[14];
  const int*   atoms     = (const int*)d_in[15];
  float* out = (float*)d_out;

  char* ws = (char*)d_ws;
  size_t off = 0;
  auto alloc = [&](size_t bytes) -> char* {
    char* p = ws + off;
    off = (off + bytes + 255) & ~(size_t)255;
    return p;
  };

  __bf16* adjBF = (__bf16*)alloc((size_t)NATOMS * NATOMS * 2);
  __bf16* memBF = (__bf16*)alloc((size_t)NRES * NATOMS * 2);
  __bf16* linWT = (__bf16*)alloc((size_t)D128 * D128 * 2);
  __bf16* wihT  = (__bf16*)alloc((size_t)D128 * 384 * 2);
  __bf16* whhT  = (__bf16*)alloc((size_t)D128 * 384 * 2);
  float*  H     = (float*) alloc((size_t)NATOMS * D128 * 4);
  __bf16* Hbf   = (__bf16*)alloc((size_t)NATOMS * D128 * 2);
  float*  Tpart = (float*) alloc((size_t)KSLICES * NATOMS * D128 * 4);
  __bf16* Tbf   = (__bf16*)alloc((size_t)NATOMS * D128 * 2);
  float*  X     = (float*) alloc((size_t)NATOMS * D128 * 4);
  __bf16* Xbf   = (__bf16*)alloc((size_t)NATOMS * D128 * 2);
  float*  GI    = (float*) alloc((size_t)NATOMS * 384 * 4);
  float*  GH    = (float*) alloc((size_t)NATOMS * 384 * 4);
  float*  HF    = (float*) alloc((size_t)NRES * D128 * 4);
  float*  P1    = (float*) alloc((size_t)9 * 4 * 128 * 16 * 4);
  float*  Q1    = (float*) alloc((size_t)9 * 16 * 4);
  float*  C1    = (float*) alloc((size_t)15 * NRES * NRES * 4);
  float*  C2    = (float*) alloc((size_t)12 * NRES * NRES * 4);
  float*  C3    = (float*) alloc((size_t)12 * NRES * NRES * 4);

  auto grid1 = [](int n) { return dim3((unsigned)((n + 255) / 256)); };

  // One-time bf16 conversions / transposes (adjacency bf16 = 33.5MB, L2-resident)
  f2bf_kernel<<<grid1(NATOMS * NATOMS), 256, 0, stream>>>(adjacency, adjBF, NATOMS * NATOMS);
  f2bf_kernel<<<grid1(NRES * NATOMS), 256, 0, stream>>>(membership, memBF, NRES * NATOMS);
  tconv_kernel<<<grid1(D128 * D128), 256, 0, stream>>>(lin_w, linWT, D128, D128);
  tconv_kernel<<<grid1(384 * D128), 256, 0, stream>>>(w_ih, wihT, 384, D128);
  tconv_kernel<<<grid1(384 * D128), 256, 0, stream>>>(w_hh, whhT, 384, D128);
  embed_kernel<<<grid1(NATOMS * D128), 256, 0, stream>>>(emb, atoms, H, Hbf);

  auto gemm = [&](const __bf16* A, const __bf16* B, const float* bias, float* C,
                  int M, int N, int K) {
    int tiles = (M >> 4) * (N >> 4);
    wmma_gemm_kernel<<<dim3((unsigned)((tiles + 7) / 8)), 256, 0, stream>>>(A, B, bias, C, M, N, K);
  };

  // 4 graph-GRU steps
  for (int s = 0; s < 4; ++s) {
    // T = adjacency @ h : split-K strip GEMM (double-buffered TDM B panels)
    wmma_strip_gemm_kernel<<<dim3(NATOMS / (16 * 8), KSLICES), 256, 0, stream>>>(
        adjBF, Hbf, Tpart, NATOMS, NATOMS);
    reduce4_bf_kernel<<<grid1(NATOMS * D128), 256, 0, stream>>>(Tpart, Tbf, NATOMS * D128);
    gemm(Tbf, linWT, lin_b, X, NATOMS, D128, D128);        // x_in = T @ lin_w^T + b
    f2bf_kernel<<<grid1(NATOMS * D128), 256, 0, stream>>>(X, Xbf, NATOMS * D128);
    gemm(Xbf, wihT, b_ih, GI, NATOMS, 384, D128);          // gi = x_in @ w_ih^T + b_ih
    gemm(Hbf, whhT, b_hh, GH, NATOMS, 384, D128);          // gh = h @ w_hh^T + b_hh
    gru_kernel<<<grid1(NATOMS * D128), 256, 0, stream>>>(GI, GH, H, Hbf);
  }

  // hf = membership @ h  (512 x 128)
  gemm(memBF, Hbf, nullptr, HF, NRES, D128, NATOMS);

  // conv1 via exact P/Q factorization (64x fewer FLOPs than direct conv)
  conv1_pq_kernel<<<grid1(9 * 4 * 128 * 16), 256, 0, stream>>>(HF, c1w, P1, Q1);
  conv1_fast_kernel<<<dim3(32, 32), 256, 0, stream>>>(HF, P1, Q1, c1b, C1);
  convK_kernel<15, 12, true ><<<dim3(32, 32), 256, 0, stream>>>(C1, c2w, c2b, C2);
  convK_kernel<12, 12, false><<<dim3(32, 32), 256, 0, stream>>>(C2, c3w, c3b, C3);
  sym_sigmoid_kernel<<<grid1(12 * NRES * NRES), 256, 0, stream>>>(C3, out, 12 * NRES * NRES);
}
